// _EntAttMean_26680336843200
// MI455X (gfx1250) — compile-verified, matrix operand order checked by reading, not run on hardware
//
#include <hip/hip_runtime.h>
#include <hip/hip_bf16.h>

// Problem constants (from reference): B=8, S=512, E=32, D_ENT=256, D_HID=512, D_ATT=200
#define BB 8
#define SS 512
#define EE 32
#define DD 256
#define HH 512
#define AA 200
#define AP 208          // D_ATT padded to 13 * 16
#define NT 13           // N tiles of 16
#define KS 8            // K steps of 32 (256 / 32)
#define NW 4            // waves (=s values) per block

typedef __attribute__((ext_vector_type(16))) __bf16          v16bf;
typedef __attribute__((ext_vector_type(8)))  float           v8f;
typedef __attribute__((ext_vector_type(8)))  unsigned short  us8;
typedef __attribute__((ext_vector_type(16))) unsigned short  us16;
typedef __attribute__((ext_vector_type(4)))  unsigned short  us4;

__device__ __forceinline__ unsigned short f2bf(float f) {
    unsigned int u = __float_as_uint(f);
    u += 0x7FFFu + ((u >> 16) & 1u);   // round-to-nearest-even
    return (unsigned short)(u >> 16);
}

// CDNA5 has a hardware TANH transcendental (v_tanh_f32); it co-executes with
// the WMMA pipe (tracked as TRANS32), so the epilogue hides under the GEMM.
__device__ __forceinline__ float fast_tanh(float x) {
#if __has_builtin(__builtin_amdgcn_tanhf)
    return __builtin_amdgcn_tanhf(x);
#elif __has_builtin(__builtin_amdgcn_tanh_f32)
    return __builtin_amdgcn_tanh_f32(x);
#else
    return tanhf(x);
#endif
}

// ---------------------------------------------------------------------------
// Kernel 1: cmean = mean_S(cxt); h_att[b][a] = cmean[b] . Wh[:,a]  (padded to 208)
// ---------------------------------------------------------------------------
__global__ __launch_bounds__(256) void prep_hatt(const float* __restrict__ cxt,
                                                 const float* __restrict__ Wh,
                                                 float* __restrict__ h_att) {
    __shared__ float cmean[HH];
    const int b = blockIdx.x, tid = threadIdx.x;
    for (int h = tid; h < HH; h += 256) {
        float sum = 0.0f;
        const float* p = cxt + (size_t)b * SS * HH + h;
        for (int s = 0; s < SS; ++s) sum += p[(size_t)s * HH];
        cmean[h] = sum * (1.0f / (float)SS);
    }
    __syncthreads();
    if (tid < AP) {
        float dot = 0.0f;
        if (tid < AA) {
            for (int h = 0; h < HH; ++h) dot += cmean[h] * Wh[h * AA + tid];
        }
        h_att[b * AP + tid] = dot;
    }
}

// ---------------------------------------------------------------------------
// Kernel 2: repack We (256x200 f32) into per-lane bf16 B-fragments for
// v_wmma_f32_16x16x32_bf16, layout [kstep][ntile][lane][16], pad N to 208.
// Also builds zero-padded Ws (208).
// ---------------------------------------------------------------------------
__global__ __launch_bounds__(256) void prep_wefrag(const float* __restrict__ We,
                                                   const float* __restrict__ Ws,
                                                   unsigned short* __restrict__ wefrag,
                                                   float* __restrict__ ws_pad) {
    const int id = blockIdx.x * 256 + threadIdx.x;
    if (id < KS * NT * 32 * 16) {
        const int j    = id & 15;
        const int lane = (id >> 4) & 31;
        const int rest = id >> 9;
        const int n    = rest % NT;
        const int k    = rest / NT;
        const int half = lane >> 4;
        const int col  = lane & 15;
        // same K pattern as the 16-bit A layout, mirrored for B columns
        const int kk = k * 32 + ((j < 8) ? (half * 8 + j) : (16 + half * 8 + (j - 8)));
        const int N  = n * 16 + col;
        const float v = (N < AA) ? We[kk * AA + N] : 0.0f;
        wefrag[id] = f2bf(v);
    }
    if (blockIdx.x == 0 && threadIdx.x < AP)
        ws_pad[threadIdx.x] = (threadIdx.x < AA) ? Ws[threadIdx.x] : 0.0f;
}

// ---------------------------------------------------------------------------
// Main fused kernel: one wave per (b,s) tile.
//   e_att = ent_tile(32x256) @ We(256x208)  via WMMA bf16
//   logits = sum_a tanh(h + e_att) * Ws ; softmax over E=32 (one entity/lane)
//   out    = softmax . ent_tile  (f32 from global, L2-hot second pass)
// ---------------------------------------------------------------------------
__device__ __forceinline__ v16bf load_afrag(const unsigned short* p) {
    us8 lo = *(const us8*)(p);
    us8 hi = *(const us8*)(p + 16);
    us16 u = __builtin_shufflevector(lo, hi, 0,1,2,3,4,5,6,7,8,9,10,11,12,13,14,15);
    return __builtin_bit_cast(v16bf, u);
}

__global__ __launch_bounds__(NW * 32) void ent_att_main(
        const float* __restrict__ ent,
        const float* __restrict__ h_att,
        const float* __restrict__ ws_pad,
        const unsigned short* __restrict__ wefrag,
        float* __restrict__ out) {
    __shared__ unsigned short entA[NW][EE * DD];   // bf16 tiles, 16 KB per wave
    __shared__ float h_lds[AP];
    __shared__ float w_lds[AP];
    __shared__ float logits[NW][EE];
    __shared__ float wts[NW][EE];

    const int tid  = threadIdx.x;
    const int w    = tid >> 5;
    const int lane = tid & 31;
    const int b    = blockIdx.x;
    const int s    = blockIdx.y * NW + w;

    for (int i = tid; i < AP; i += NW * 32) {
        h_lds[i] = h_att[b * AP + i];
        w_lds[i] = ws_pad[i];
    }

    // ---- stage ent tile (32x256 f32) -> bf16 LDS, row-major [e][d] ----
    const float* tile = ent + (size_t)(b * SS + s) * EE * DD;
    {
        const float4* t4 = (const float4*)tile;
        unsigned short* myA = &entA[w][0];
        for (int i = lane; i < (EE * DD) / 4; i += 32) {
            float4 v = t4[i];
            us4 h4;
            h4.x = f2bf(v.x); h4.y = f2bf(v.y); h4.z = f2bf(v.z); h4.w = f2bf(v.w);
            *(us4*)(myA + i * 4) = h4;
        }
    }
    __syncthreads();

    // ---- GEMM + tanh + dot(Ws) fused epilogue ----
    const int col   = lane & 15;
    const int half8 = (lane >> 4) * 8;
    const unsigned short* row0 = &entA[w][col * DD];        // mtile 0: e = col
    const unsigned short* row1 = &entA[w][(16 + col) * DD]; // mtile 1: e = 16+col

    float p0[8], p1[8];
#pragma unroll
    for (int r = 0; r < 8; ++r) { p0[r] = 0.0f; p1[r] = 0.0f; }

    for (int nt = 0; nt < NT; ++nt) {
        v8f acc0 = {0,0,0,0,0,0,0,0};
        v8f acc1 = {0,0,0,0,0,0,0,0};
#pragma unroll
        for (int k = 0; k < KS; ++k) {
            const int d0 = k * 32 + half8;
            v16bf a0 = load_afrag(row0 + d0);
            v16bf a1 = load_afrag(row1 + d0);
            us16 bu = *(const us16*)(wefrag + (((k * NT + nt) * 32 + lane) << 4));
            v16bf bf = __builtin_bit_cast(v16bf, bu);
            acc0 = __builtin_amdgcn_wmma_f32_16x16x32_bf16(false, a0, false, bf,
                                                           (short)0, acc0, false, false);
            acc1 = __builtin_amdgcn_wmma_f32_16x16x32_bf16(false, a1, false, bf,
                                                           (short)0, acc1, false, false);
        }
        // C layout: lane, VGPR r -> (M = r + 8*(lane>>4), N = nt*16 + (lane&15))
        const float hv = h_lds[nt * 16 + col];
        const float wv = w_lds[nt * 16 + col];
#pragma unroll
        for (int r = 0; r < 8; ++r) {
            p0[r] += fast_tanh(hv + acc0[r]) * wv;
            p1[r] += fast_tanh(hv + acc1[r]) * wv;
        }
    }

    // reduce partial logits across the 16 lanes of each half (N classes)
#pragma unroll
    for (int r = 0; r < 8; ++r) {
#pragma unroll
        for (int m = 1; m <= 8; m <<= 1) {
            p0[r] += __shfl_xor(p0[r], m);
            p1[r] += __shfl_xor(p1[r], m);
        }
    }
    if (col == 0) {
        const int half = lane >> 4;
#pragma unroll
        for (int r = 0; r < 8; ++r) {
            logits[w][8 * half + r]      = p0[r];  // entities  0..15
            logits[w][16 + 8 * half + r] = p1[r];  // entities 16..31
        }
    }
    __syncthreads();

    // ---- softmax over E=32: one entity per lane ----
    float lg = logits[w][lane];
    float mx = lg;
#pragma unroll
    for (int m = 16; m >= 1; m >>= 1) mx = fmaxf(mx, __shfl_xor(mx, m));
    float ex = __expf(lg - mx);
    float sm = ex;
#pragma unroll
    for (int m = 16; m >= 1; m >>= 1) sm += __shfl_xor(sm, m);
    wts[w][lane] = ex / sm;
    __syncthreads();

    // ---- weighted sum over entities, f32 ent from global (L2-hot) ----
    const float4* t4 = (const float4*)tile;
    float4* o4 = (float4*)(out + (size_t)(b * SS + s) * DD);
#pragma unroll
    for (int c = 0; c < 2; ++c) {
        const int d4 = c * 32 + lane;          // float4 index: d = d4*4
        float4 acc = make_float4(0.f, 0.f, 0.f, 0.f);
        for (int e = 0; e < EE; ++e) {
            const float we = wts[w][e];
            float4 v = t4[e * (DD / 4) + d4];
            acc.x += we * v.x; acc.y += we * v.y;
            acc.z += we * v.z; acc.w += we * v.w;
        }
        o4[d4] = acc;
    }
}

// ---------------------------------------------------------------------------
extern "C" void kernel_launch(void* const* d_in, const int* in_sizes, int n_in,
                              void* d_out, int out_size, void* d_ws, size_t ws_size,
                              hipStream_t stream) {
    const float* cxt = (const float*)d_in[0];
    const float* ent = (const float*)d_in[1];
    const float* We  = (const float*)d_in[2];
    const float* Wh  = (const float*)d_in[3];
    const float* Ws  = (const float*)d_in[4];
    float* out = (float*)d_out;

    // workspace layout
    float* h_att  = (float*)d_ws;                        // 8*208 f32
    float* ws_pad = h_att + BB * AP;                     // 208 f32
    size_t frag_off = (((size_t)(BB * AP + AP) * sizeof(float)) + 127) & ~(size_t)127;
    unsigned short* wefrag = (unsigned short*)((char*)d_ws + frag_off);

    prep_hatt<<<dim3(BB), dim3(256), 0, stream>>>(cxt, Wh, h_att);

    const int nfrag = KS * NT * 32 * 16;
    prep_wefrag<<<dim3((nfrag + 255) / 256), dim3(256), 0, stream>>>(We, Ws, wefrag, ws_pad);

    ent_att_main<<<dim3(BB, SS / NW), dim3(NW * 32), 0, stream>>>(ent, h_att, ws_pad,
                                                                  wefrag, out);
}
